// MultiHeadAttention_40604620816403
// MI455X (gfx1250) — matrix operand order repaired and checked by reading, not compiled, not run on hardware
//
#include <hip/hip_runtime.h>
#include <stdint.h>

// Problem constants (fixed by the reference): B=4, N=4096, D=1024, H=16, DK=64
#define MTOK   16384          // B*N tokens
#define DDIM   1024           // model dim
#define KDIM   1024           // GEMM reduction dim
#define NDIM   1024           // GEMM output dim

#define BM 128
#define BN 128
#define BK 64
#define NKB (KDIM / BK)       // 16 k-blocks
#define LSTRIDE 72            // LDS row stride in halves: 144B, 16B-aligned, conflict-free

typedef __attribute__((ext_vector_type(16))) __bf16          v16bf;
typedef __attribute__((ext_vector_type(8)))  float           v8f;
typedef __attribute__((ext_vector_type(16))) unsigned short  v16us;
typedef __attribute__((ext_vector_type(4)))  float           float4v;
typedef __attribute__((ext_vector_type(4)))  unsigned int    uint4v;
typedef __attribute__((ext_vector_type(4)))  unsigned short  ushort4v;
typedef __attribute__((ext_vector_type(4)))  int             int4v;

union Frag {
    v16bf  b;
    v16us  u;
    uint4v q[2];
};

#if defined(__has_builtin)
#if __has_builtin(__builtin_amdgcn_global_load_async_to_lds_b128)
#define HAVE_ASYNC_LDS 1
#endif
#endif

#if defined(HAVE_ASYNC_LDS)
template <int OFF>
__device__ __forceinline__ void async_copy_b128(const unsigned short* g,
                                                unsigned short* l) {
    __builtin_amdgcn_global_load_async_to_lds_b128(
        (__attribute__((address_space(1))) int4v*)(g),
        (__attribute__((address_space(3))) int4v*)(l),
        OFF, 0);
}
#if __has_builtin(__builtin_amdgcn_s_wait_asynccnt)
#define ASYNC_WAIT(n) __builtin_amdgcn_s_wait_asynccnt(n)
#else
#define ASYNC_WAIT(n) asm volatile("s_wait_asynccnt %0" ::"i"(n) : "memory")
#endif
#endif

__device__ __forceinline__ unsigned short f32_to_bf16(float f) {
    unsigned int u = __float_as_uint(f);
    unsigned int r = u + 0x7FFFu + ((u >> 16) & 1u);   // round-to-nearest-even
    return (unsigned short)(r >> 16);
}

// ---------------------------------------------------------------------------
// fp32 -> bf16 bulk conversion (4 elements/thread)
// ---------------------------------------------------------------------------
__global__ __launch_bounds__(256)
void cvt_f32_bf16(const float* __restrict__ in, unsigned short* __restrict__ out, int n4) {
    int i = blockIdx.x * blockDim.x + threadIdx.x;
    if (i >= n4) return;
    float4v v = *((const float4v*)in + i);
    ushort4v o;
    #pragma unroll
    for (int j = 0; j < 4; ++j) o[j] = f32_to_bf16(v[j]);
    *((ushort4v*)out + i) = o;
}

// ---------------------------------------------------------------------------
// GEMM: C[m,n] = sum_k A[m,k] * W[n,k] + bias[n]
//   A: MTOK x KDIM bf16, W: NDIM x KDIM bf16 (row-major = B^T).
//   256 threads = 8 waves; block tile 128x128; wave tile 32x64 (2x4 WMMA tiles).
//   A/B 128x64 sub-tiles staged in LDS; async-to-LDS double buffering when
//   the gfx1250 builtin is available, register+ds_store staging otherwise.
//   grid.z selects weight/bias/output set (fused QKV).
// ---------------------------------------------------------------------------
template <bool OUT_BF16>
__global__ __launch_bounds__(256)
void gemm_bf16(const unsigned short* __restrict__ A,
               const unsigned short* __restrict__ Wbase,
               const float* __restrict__ b0, const float* __restrict__ b1,
               const float* __restrict__ b2,
               unsigned short* __restrict__ outb, float* __restrict__ outf)
{
#if defined(HAVE_ASYNC_LDS)
    __shared__ unsigned short sAB[2][2][BM * LSTRIDE];   // [buf][A|B][...] 72KB
#else
    __shared__ unsigned short sAB[1][2][BM * LSTRIDE];   // single buffer     36KB
#endif

    const int z = blockIdx.z;
    const unsigned short* W = Wbase + (size_t)z * (KDIM * NDIM);
    const float* bias = (z == 0) ? b0 : (z == 1) ? b1 : b2;

    const int tid  = threadIdx.x;
    const int lane = tid & 31;
    const int wid  = tid >> 5;
    const int l15  = lane & 15;
    const int hi   = lane >> 4;

    const int bMblk = blockIdx.y * BM;
    const int bNblk = blockIdx.x * BN;
    const int wm = (wid & 3) * 32;    // wave M origin within block tile
    const int wn = (wid >> 2) * 64;   // wave N origin within block tile

    // 16-bit A fragment K-chunks per lane half: {0..7,16..23} / {8..15,24..31}
    const int ka = hi ? 8 : 0;
    // 16-bit B fragment K-chunks per lane half: {0..7,16..23} / {8..15,24..31}
    const int kw = hi ? 8 : 0;

    // Cooperative staging map: thread -> (row, 32-half chunk) of the 128x64 tile
    const int srow = tid >> 1;
    const int soff = (tid & 1) * 32;
    const unsigned short* gA = A + (size_t)(bMblk + srow) * KDIM + soff;
    const unsigned short* gW = W + (size_t)(bNblk + srow) * KDIM + soff;

    v8f acc[2][4];
    #pragma unroll
    for (int mi = 0; mi < 2; ++mi)
        #pragma unroll
        for (int ni = 0; ni < 4; ++ni)
            acc[mi][ni] = (v8f){0.f, 0.f, 0.f, 0.f, 0.f, 0.f, 0.f, 0.f};

    auto compute = [&](int buf) {
        const unsigned short* bA = &sAB[buf][0][0];
        const unsigned short* bB = &sAB[buf][1][0];
        #pragma unroll
        for (int ks = 0; ks < BK; ks += 32) {
            Frag a0, a1, bfr[4];
            const unsigned short* r0 = bA + (wm + l15) * LSTRIDE + ks;
            a0.q[0] = *(const uint4v*)(r0 + ka);
            a0.q[1] = *(const uint4v*)(r0 + ka + 16);
            const unsigned short* r1 = r0 + 16 * LSTRIDE;
            a1.q[0] = *(const uint4v*)(r1 + ka);
            a1.q[1] = *(const uint4v*)(r1 + ka + 16);
            #pragma unroll
            for (int ni = 0; ni < 4; ++ni) {
                const unsigned short* rb = bB + (wn + ni * 16 + l15) * LSTRIDE + ks;
                bfr[ni].q[0] = *(const uint4v*)(rb + kw);
                bfr[ni].q[1] = *(const uint4v*)(rb + kw + 16);
            }
            #pragma unroll
            for (int ni = 0; ni < 4; ++ni) {
                acc[0][ni] = __builtin_amdgcn_wmma_f32_16x16x32_bf16(
                    false, a0.b, false, bfr[ni].b, (short)0, acc[0][ni], false, false);
                acc[1][ni] = __builtin_amdgcn_wmma_f32_16x16x32_bf16(
                    false, a1.b, false, bfr[ni].b, (short)0, acc[1][ni], false, false);
            }
        }
    };

#if defined(HAVE_ASYNC_LDS)
    auto stage = [&](int kb, int buf) {
        const unsigned short* ga = gA + kb * BK;
        const unsigned short* gw = gW + kb * BK;
        unsigned short* la = &sAB[buf][0][srow * LSTRIDE + soff];
        unsigned short* lb = &sAB[buf][1][srow * LSTRIDE + soff];
        async_copy_b128<0>(ga, la);
        async_copy_b128<16>(ga, la);
        async_copy_b128<32>(ga, la);
        async_copy_b128<48>(ga, la);
        async_copy_b128<0>(gw, lb);
        async_copy_b128<16>(gw, lb);
        async_copy_b128<32>(gw, lb);
        async_copy_b128<48>(gw, lb);
    };

    stage(0, 0);
    for (int kb = 0; kb < NKB; ++kb) {
        const int cur = kb & 1;
        if (kb + 1 < NKB) {
            stage(kb + 1, cur ^ 1);
            ASYNC_WAIT(8);            // drain only the current tile's 8 copies
        } else {
            ASYNC_WAIT(0);
        }
        __syncthreads();              // all waves' staging of `cur` visible
        compute(cur);
        __syncthreads();              // `cur` free for reuse
    }
#else
    for (int kb = 0; kb < NKB; ++kb) {
        uint4v ra[4], rw[4];
        #pragma unroll
        for (int c = 0; c < 4; ++c) {
            ra[c] = *(const uint4v*)(gA + kb * BK + c * 8);
            rw[c] = *(const uint4v*)(gW + kb * BK + c * 8);
        }
        __syncthreads();              // previous compute done, buffer free
        unsigned short* la = &sAB[0][0][srow * LSTRIDE + soff];
        unsigned short* lb = &sAB[0][1][srow * LSTRIDE + soff];
        #pragma unroll
        for (int c = 0; c < 4; ++c) {
            *(uint4v*)(la + c * 8) = ra[c];
            *(uint4v*)(lb + c * 8) = rw[c];
        }
        __syncthreads();              // staging visible to all waves
        compute(0);
    }
#endif

    // Epilogue: C VGPR j -> row (j + 8*hi), col l15 within each 16x16 tile
    #pragma unroll
    for (int ni = 0; ni < 4; ++ni) {
        const int col = bNblk + wn + ni * 16 + l15;
        const float bb = bias[col];
        #pragma unroll
        for (int mi = 0; mi < 2; ++mi) {
            #pragma unroll
            for (int j = 0; j < 8; ++j) {
                const int row = bMblk + wm + mi * 16 + j + hi * 8;
                const float v = acc[mi][ni][j] + bb;
                if (OUT_BF16)
                    outb[(size_t)z * (MTOK * (size_t)NDIM) + (size_t)row * NDIM + col] =
                        f32_to_bf16(v);
                else
                    outf[(size_t)row * NDIM + col] = v;
            }
        }
    }
}

// ---------------------------------------------------------------------------
// Per-token head-vs-head attention: one wave per token.
//   scores(16x16) = Q(16x64) @ K^T  -> 2x WMMA bf16 k=32
//   mask + softmax over last axis (lanes 0-15 own one row each, via LDS)
//   out(16x64) = attn(16x16, zero-padded to K=32) @ V -> 4x WMMA
// ---------------------------------------------------------------------------
__global__ __launch_bounds__(256)
void attn_kernel(const unsigned short* __restrict__ Qb,
                 const unsigned short* __restrict__ Kb,
                 const unsigned short* __restrict__ Vb,
                 const int* __restrict__ mask,
                 unsigned short* __restrict__ Ob)
{
    __shared__ float s_sc[8][256];
    const int lane = threadIdx.x & 31;
    const int w    = threadIdx.x >> 5;
    const size_t t = (size_t)blockIdx.x * 8 + w;   // token index
    const int l15  = lane & 15;
    const int hi   = lane >> 4;
    const int ka   = hi ? 8 : 0;
    const int kw   = hi ? 8 : 0;

    const unsigned short* Qt = Qb + t * 1024 + l15 * 64;  // row h = l15 of Q_tok
    const unsigned short* Kt = Kb + t * 1024 + l15 * 64;  // row g = l15 of K_tok

    v8f sacc = (v8f){0.f, 0.f, 0.f, 0.f, 0.f, 0.f, 0.f, 0.f};
    #pragma unroll
    for (int k0 = 0; k0 < 64; k0 += 32) {
        Frag qa, kb;
        qa.q[0] = *(const uint4v*)(Qt + k0 + ka);
        qa.q[1] = *(const uint4v*)(Qt + k0 + ka + 16);
        kb.q[0] = *(const uint4v*)(Kt + k0 + kw);
        kb.q[1] = *(const uint4v*)(Kt + k0 + kw + 16);
        sacc = __builtin_amdgcn_wmma_f32_16x16x32_bf16(
            false, qa.b, false, kb.b, (short)0, sacc, false, false);
    }

    // scores -> LDS (scaled by 1/sqrt(64))
    #pragma unroll
    for (int j = 0; j < 8; ++j)
        s_sc[w][(j + hi * 8) * 16 + l15] = sacc[j] * 0.125f;
    __syncthreads();

    // masked softmax over last axis; lanes 0-15 each own one row h
    if (lane < 16) {
        const int h = lane;
        const int* mrow = mask + t * 256 + h * 16;
        float vals[16];
        float mx = -3.0e38f;
        #pragma unroll
        for (int g = 0; g < 16; ++g) {
            float x = s_sc[w][h * 16 + g];
            if (mrow[g] == 0) x = -1.0e9f;
            vals[g] = x;
            mx = fmaxf(mx, x);
        }
        float s = 0.f;
        #pragma unroll
        for (int g = 0; g < 16; ++g) {
            float e = __expf(vals[g] - mx);
            vals[g] = e;
            s += e;
        }
        const float inv = 1.0f / s;
        #pragma unroll
        for (int g = 0; g < 16; ++g)
            s_sc[w][h * 16 + g] = vals[g] * inv;
    }
    __syncthreads();

    // attn as A fragment: row h=l15, K = g in {0..7 | 8..15} per lane half, pad to K=32
    Frag af;
    #pragma unroll
    for (int i = 0; i < 8; ++i)
        af.u[i] = f32_to_bf16(s_sc[w][l15 * 16 + hi * 8 + i]);
    #pragma unroll
    for (int i = 8; i < 16; ++i) af.u[i] = (unsigned short)0;

    const unsigned short* Vt = Vb + t * 1024;
    unsigned short*       Ot = Ob + t * 1024;

    #pragma unroll
    for (int dt = 0; dt < 4; ++dt) {
        // B fragment: B[g,d]=V[g,d]; elems 0-7 hold g = 8*hi + i, elems 8-15 pad=0
        Frag vf;
        #pragma unroll
        for (int i = 0; i < 16; ++i) vf.u[i] = (unsigned short)0;
        {
            const int d = dt * 16 + l15;
            #pragma unroll
            for (int i = 0; i < 8; ++i) vf.u[i] = Vt[(hi * 8 + i) * 64 + d];
        }
        v8f o = (v8f){0.f, 0.f, 0.f, 0.f, 0.f, 0.f, 0.f, 0.f};
        o = __builtin_amdgcn_wmma_f32_16x16x32_bf16(
            false, af.b, false, vf.b, (short)0, o, false, false);
        #pragma unroll
        for (int j = 0; j < 8; ++j)
            Ot[(j + hi * 8) * 64 + dt * 16 + l15] = f32_to_bf16(o[j]);
    }
}

// ---------------------------------------------------------------------------
// Host-side orchestration
// ---------------------------------------------------------------------------
extern "C" void kernel_launch(void* const* d_in, const int* in_sizes, int n_in,
                              void* d_out, int out_size, void* d_ws, size_t ws_size,
                              hipStream_t stream) {
    const float* x    = (const float*)d_in[0];
    const float* Wq   = (const float*)d_in[1];
    const float* bq   = (const float*)d_in[2];
    const float* Wk   = (const float*)d_in[3];
    const float* bk   = (const float*)d_in[4];
    const float* Wv   = (const float*)d_in[5];
    const float* bv   = (const float*)d_in[6];
    const float* Wo   = (const float*)d_in[7];
    const float* bo   = (const float*)d_in[8];
    const int*   mask = (const int*)d_in[10];
    float*       out  = (float*)d_out;

    // Workspace layout (bytes):
    //   [0, 32M)        xb   : x in bf16  (reused as attn-out bf16 after QKV)
    //   [32M, 38M)      Wqkv : Wq,Wk,Wv bf16 (contiguous)
    //   [38M, 40M)      Wob  : Wo bf16
    //   [40M, 136M)     QKVb : Q,K,V bf16 (contiguous)
    char* ws = (char*)d_ws;
    unsigned short* xb    = (unsigned short*)(ws);
    unsigned short* Wqkvb = (unsigned short*)(ws + 33554432);
    unsigned short* Wob   = (unsigned short*)(ws + 39845888);
    unsigned short* QKVb  = (unsigned short*)(ws + 41943040);

    const size_t TOKELEMS = (size_t)MTOK * DDIM;   // 16,777,216
    const size_t WELEMS   = (size_t)DDIM * DDIM;   // 1,048,576

    // 1) fp32 -> bf16 conversions (x, Wq, Wk, Wv, Wo)
    {
        int n4 = (int)(TOKELEMS / 4);
        cvt_f32_bf16<<<(n4 + 255) / 256, 256, 0, stream>>>(x, xb, n4);
        int w4 = (int)(WELEMS / 4);
        cvt_f32_bf16<<<(w4 + 255) / 256, 256, 0, stream>>>(Wq, Wqkvb,              w4);
        cvt_f32_bf16<<<(w4 + 255) / 256, 256, 0, stream>>>(Wk, Wqkvb + WELEMS,     w4);
        cvt_f32_bf16<<<(w4 + 255) / 256, 256, 0, stream>>>(Wv, Wqkvb + 2 * WELEMS, w4);
        cvt_f32_bf16<<<(w4 + 255) / 256, 256, 0, stream>>>(Wo, Wob,                w4);
    }

    // 2) fused Q/K/V projections: grid.z = 3 selects weight/bias/output set
    gemm_bf16<true><<<dim3(NDIM / BN, MTOK / BM, 3), 256, 0, stream>>>(
        xb, Wqkvb, bq, bk, bv, QKVb, nullptr);

    // 3) per-token attention (writes bf16 attn-out over xb)
    attn_kernel<<<MTOK / 8, 256, 0, stream>>>(
        QKVb, QKVb + TOKELEMS, QKVb + 2 * TOKELEMS, mask, xb);

    // 4) output projection -> f32 d_out
    gemm_bf16<false><<<dim3(NDIM / BN, MTOK / BM, 1), 256, 0, stream>>>(
        xb, Wob, bo, bo, bo, nullptr, out);
}
